// MultiLayersGAT_74491912782358
// MI455X (gfx1250) — compile-verified, matrix operand order checked by reading, not run on hardware
//
#include <hip/hip_runtime.h>
#include <hip/hip_bf16.h>
#include <math.h>

#define NNODES 170000
#define NEDGES 1200000
#define INDIM  128
#define EPSBN  1e-5f
#define SLOPE  0.2f

typedef __attribute__((ext_vector_type(16))) __bf16 v16bf;
typedef __attribute__((ext_vector_type(8)))  float  v8f;

__device__ __forceinline__ unsigned short f2bf(float f) {
  unsigned int u = __float_as_uint(f);
  u += 0x7fffu + ((u >> 16) & 1u);           // round-to-nearest-even
  return (unsigned short)(u >> 16);
}

// float atomic max via int punning (valid across sign mix; init = -inf)
__device__ __forceinline__ void atomicMaxFloat(float* addr, float val) {
  if (val >= 0.0f) {
    atomicMax((int*)addr, __float_as_int(val));
  } else {
    atomicMin((unsigned int*)addr, __float_as_uint(val));
  }
}

// ---------------------------------------------------------------------------
// C[M,Nout] = A[M,Kdim] @ B[Kdim,Nout]; bf16 WMMA, f32 accumulate.
// Block: 256 threads = 8 waves; block tile 64x128; wave tile 16x64
// (4 accumulators sharing one A operand -> 4 back-to-back WMMAs); K step 32.
// ---------------------------------------------------------------------------
__global__ __launch_bounds__(256)
void gemm_wmma_bf16(const float* __restrict__ A, const float* __restrict__ B,
                    float* __restrict__ C, int M, int Kdim, int Nout) {
  __shared__ __align__(16) unsigned short As[64 * 32];    // [m][k]
  __shared__ __align__(16) unsigned short Bs[128 * 32];   // [n][k] (transposed)

  const int tid  = threadIdx.x;
  const int lane = tid & 31;
  const int wave = tid >> 5;
  const int mt   = wave >> 1;          // 0..3  (M sub-tile)
  const int ng   = wave & 1;           // 0..1  (N sub-tile group of 4)
  const int row0 = blockIdx.x * 64;
  const int n0   = blockIdx.y * 128;
  const int half = lane >> 4;          // lane group 0/1
  const int mr   = lane & 15;

  v8f acc0 = {}, acc1 = {}, acc2 = {}, acc3 = {};

  for (int kk = 0; kk < Kdim; kk += 32) {
    // ---- stage A tile 64x32: branch-free clamped float4 loads ----
#pragma unroll
    for (int i = 0; i < 2; ++i) {
      int idx = tid + i * 256;                   // 0..511 float4 slots
      int r = idx >> 3, c4 = (idx & 7) * 4;
      int gr = row0 + r;
      int grc = (gr < M) ? gr : (M - 1);         // clamp, always-valid load
      float4 v = *reinterpret_cast<const float4*>(&A[(long)grc * Kdim + kk + c4]);
      bool ok = gr < M;
      ushort4 h;
      h.x = f2bf(ok ? v.x : 0.0f);
      h.y = f2bf(ok ? v.y : 0.0f);
      h.z = f2bf(ok ? v.z : 0.0f);
      h.w = f2bf(ok ? v.w : 0.0f);
      *reinterpret_cast<ushort4*>(&As[r * 32 + c4]) = h;
    }
    // ---- stage B tile transposed Bs[n][k] = B[kk+k][n0+n] ----
#pragma unroll
    for (int i = 0; i < 4; ++i) {
      int idx = tid + i * 256;                   // 0..1023 float4 slots
      int k = idx >> 5, nb = (idx & 31) * 4;
      int gb = n0 + nb;
      int gbc = (gb + 4 <= Nout) ? gb : (Nout - 4);   // Nout multiple of 4
      float4 v = *reinterpret_cast<const float4*>(&B[(long)(kk + k) * Nout + gbc]);
      bool ok = (gb + 4 <= Nout);                     // tile cols all-in or all-out
      Bs[(nb + 0) * 32 + k] = f2bf(ok ? v.x : 0.0f);
      Bs[(nb + 1) * 32 + k] = f2bf(ok ? v.y : 0.0f);
      Bs[(nb + 2) * 32 + k] = f2bf(ok ? v.z : 0.0f);
      Bs[(nb + 3) * 32 + k] = f2bf(ok ? v.w : 0.0f);
    }
    // prefetch next-K A tile (global_prefetch_b8)
    if (kk + 32 < Kdim) {
      int r = tid >> 2, c = (tid & 3) * 8;
      int gr = row0 + r;
      if (gr < M) __builtin_prefetch(&A[(long)gr * Kdim + kk + 32 + c], 0, 0);
    }
    __syncthreads();

    // Assemble documented 16-bit A/B operand layouts:
    // lanes 0-15 / 16-31 hold K halves {base, base+8}; each v16bf = two 16B chunks.
    union U { v16bf v; uint4 q[2]; };
    U va, vb0, vb1, vb2, vb3;
    const unsigned short* pa = &As[(mt * 16 + mr) * 32 + half * 8];
    va.q[0] = *reinterpret_cast<const uint4*>(pa);        // K = half*8 .. +7
    va.q[1] = *reinterpret_cast<const uint4*>(pa + 16);   // K = 16+half*8 .. +7
    const unsigned short* pb = &Bs[(ng * 64 + mr) * 32 + half * 8];
    vb0.q[0] = *reinterpret_cast<const uint4*>(pb);
    vb0.q[1] = *reinterpret_cast<const uint4*>(pb + 16);
    vb1.q[0] = *reinterpret_cast<const uint4*>(pb + 512);
    vb1.q[1] = *reinterpret_cast<const uint4*>(pb + 528);
    vb2.q[0] = *reinterpret_cast<const uint4*>(pb + 1024);
    vb2.q[1] = *reinterpret_cast<const uint4*>(pb + 1040);
    vb3.q[0] = *reinterpret_cast<const uint4*>(pb + 1536);
    vb3.q[1] = *reinterpret_cast<const uint4*>(pb + 1552);

    acc0 = __builtin_amdgcn_wmma_f32_16x16x32_bf16(false, va.v, false, vb0.v,
                                                   (short)0, acc0, false, false);
    acc1 = __builtin_amdgcn_wmma_f32_16x16x32_bf16(false, va.v, false, vb1.v,
                                                   (short)0, acc1, false, false);
    acc2 = __builtin_amdgcn_wmma_f32_16x16x32_bf16(false, va.v, false, vb2.v,
                                                   (short)0, acc2, false, false);
    acc3 = __builtin_amdgcn_wmma_f32_16x16x32_bf16(false, va.v, false, vb3.v,
                                                   (short)0, acc3, false, false);
    __syncthreads();
  }

  // D layout: VGPR r -> M = r (+8 for lanes 16-31); N = lane & 15
  const int gcol = n0 + ng * 64 + mr;
#pragma unroll
  for (int r = 0; r < 8; ++r) {
    int grow = row0 + mt * 16 + half * 8 + r;
    if (grow < M) {
      long base = (long)grow * Nout;
      if (gcol      < Nout) C[base + gcol]      = acc0[r];
      if (gcol + 16 < Nout) C[base + gcol + 16] = acc1[r];
      if (gcol + 32 < Nout) C[base + gcol + 32] = acc2[r];
      if (gcol + 48 < Nout) C[base + gcol + 48] = acc3[r];
    }
  }
}

// ---------------------------------------------------------------------------
// el[n,k] = sum_f feat[n,k,f]*al[k,f]; er likewise. One wave per node.
// ---------------------------------------------------------------------------
__global__ __launch_bounds__(256)
void attn_dots(const float* __restrict__ feat, const float* __restrict__ al,
               const float* __restrict__ ar, float* __restrict__ el,
               float* __restrict__ er, int K, int F) {
  int lane = threadIdx.x & 31;
  int node = blockIdx.x * 8 + (threadIdx.x >> 5);
  if (node >= NNODES) return;
  int KH = K * F;
  for (int k = 0; k < K; ++k) {
    float sl = 0.0f, sr = 0.0f;
    for (int f = lane; f < F; f += 32) {
      float v = feat[(long)node * KH + k * F + f];
      sl += v * al[k * F + f];
      sr += v * ar[k * F + f];
    }
#pragma unroll
    for (int m = 16; m >= 1; m >>= 1) {
      sl += __shfl_xor(sl, m, 32);
      sr += __shfl_xor(sr, m, 32);
    }
    if (lane == 0) { el[node * K + k] = sl; er[node * K + k] = sr; }
  }
}

__global__ void fill_kernel(float* __restrict__ p, float v, long n) {
  long i = (long)blockIdx.x * blockDim.x + threadIdx.x;
  if (i < n) p[i] = v;
}

// leaky-relu edge scores + running per-(dst,head) max
__global__ void edge_score(const int* __restrict__ src, const int* __restrict__ dst,
                           const float* __restrict__ el, const float* __restrict__ er,
                           float* __restrict__ sbuf, float* __restrict__ mbuf, int K) {
  long idx = (long)blockIdx.x * blockDim.x + threadIdx.x;
  if (idx >= (long)NEDGES * K) return;
  int e = (int)(idx / K);
  int k = (int)(idx - (long)e * K);
  float x = el[src[e] * K + k] + er[dst[e] * K + k];
  float s = (x >= 0.0f) ? x : SLOPE * x;
  sbuf[idx] = s;
  atomicMaxFloat(&mbuf[dst[e] * K + k], s);
}

// stable exp + per-(dst,head) denominator
__global__ void edge_exp(const int* __restrict__ dst, const float* __restrict__ mbuf,
                         float* __restrict__ sbuf, float* __restrict__ denom, int K) {
  long idx = (long)blockIdx.x * blockDim.x + threadIdx.x;
  if (idx >= (long)NEDGES * K) return;
  int e = (int)(idx / K);
  int k = (int)(idx - (long)e * K);
  float ex = __expf(sbuf[idx] - mbuf[dst[e] * K + k]);
  sbuf[idx] = ex;
  atomicAdd(&denom[dst[e] * K + k], ex);
}

// rst[dst] += feat[src] * alpha ; one wave per edge (coalesced row gather)
__global__ __launch_bounds__(256)
void aggregate(const int* __restrict__ src, const int* __restrict__ dst,
               const float* __restrict__ feat, const float* __restrict__ sbuf,
               const float* __restrict__ denom, float* __restrict__ rst,
               int K, int F) {
  int lane = threadIdx.x & 31;
  long e = (long)blockIdx.x * 8 + (threadIdx.x >> 5);
  if (e >= NEDGES) return;
  int s = src[e], d = dst[e];
  int KH = K * F;
  for (int k = 0; k < K; ++k) {
    float alpha = sbuf[e * K + k] / denom[d * K + k];
    for (int f = lane; f < F; f += 32) {
      atomicAdd(&rst[(long)d * KH + k * F + f],
                feat[(long)s * KH + k * F + f] * alpha);
    }
  }
}

// per-column sum / sum-of-squares (thread = column, coalesced row-major reads)
__global__ __launch_bounds__(256)
void bn_stats(const float* __restrict__ rst, float* __restrict__ bnsum,
              float* __restrict__ bnsqs, int KH, int rowsPerBlock) {
  int c = threadIdx.x;
  if (c >= KH) return;
  long r0 = (long)blockIdx.x * rowsPerBlock;
  long r1 = r0 + rowsPerBlock;
  if (r1 > NNODES) r1 = NNODES;
  float s = 0.0f, q = 0.0f;
  for (long r = r0; r < r1; ++r) {
    float v = rst[r * KH + c];
    s += v; q += v * v;
  }
  atomicAdd(&bnsum[c], s);
  atomicAdd(&bnsqs[c], q);
}

// (x - mu) * rsqrt(var + eps) * g + be, then ReLU.
// Note: layer bias b cancels exactly through BN mean subtraction -> omitted.
__global__ void bn_apply_relu(const float* __restrict__ rst,
                              const float* __restrict__ bnsum,
                              const float* __restrict__ bnsqs,
                              const float* __restrict__ g,
                              const float* __restrict__ be,
                              float* __restrict__ out, int KH) {
  long idx = (long)blockIdx.x * blockDim.x + threadIdx.x;
  if (idx >= (long)NNODES * KH) return;
  int c = (int)(idx % KH);
  const float invN = 1.0f / (float)NNODES;
  float mu  = bnsum[c] * invN;
  float var = bnsqs[c] * invN - mu * mu;
  float y = (rst[idx] - mu) * rsqrtf(var + EPSBN) * g[c] + be[c];
  out[idx] = fmaxf(y, 0.0f);
}

// row softmax over C=40; one wave per node
__global__ __launch_bounds__(256)
void softmax40(const float* __restrict__ h, float* __restrict__ out) {
  const int C = 40;
  int lane = threadIdx.x & 31;
  int node = blockIdx.x * 8 + (threadIdx.x >> 5);
  if (node >= NNODES) return;
  float v0 = h[(long)node * C + lane];
  float v1 = (lane + 32 < C) ? h[(long)node * C + lane + 32] : -INFINITY;
  float m = fmaxf(v0, v1);
#pragma unroll
  for (int s = 16; s >= 1; s >>= 1) m = fmaxf(m, __shfl_xor(m, s, 32));
  float e0 = __expf(v0 - m);
  float e1 = (lane + 32 < C) ? __expf(v1 - m) : 0.0f;
  float sum = e0 + e1;
#pragma unroll
  for (int s = 16; s >= 1; s >>= 1) sum += __shfl_xor(sum, s, 32);
  float inv = 1.0f / sum;
  out[(long)node * C + lane] = e0 * inv;
  if (lane + 32 < C) out[(long)node * C + lane + 32] = e1 * inv;
}

extern "C" void kernel_launch(void* const* d_in, const int* in_sizes, int n_in,
                              void* d_out, int out_size, void* d_ws, size_t ws_size,
                              hipStream_t stream) {
  const float* x   = (const float*)d_in[0];
  const int*   src = (const int*)d_in[1];
  const int*   dst = (const int*)d_in[2];
  const float* W[3]  = {(const float*)d_in[3],  (const float*)d_in[9],  (const float*)d_in[15]};
  const float* al[3] = {(const float*)d_in[4],  (const float*)d_in[10], (const float*)d_in[16]};
  const float* ar[3] = {(const float*)d_in[5],  (const float*)d_in[11], (const float*)d_in[17]};
  const float* g[3]  = {(const float*)d_in[7],  (const float*)d_in[13], (const float*)d_in[19]};
  const float* be[3] = {(const float*)d_in[8],  (const float*)d_in[14], (const float*)d_in[20]};
  // b0/b1/b2 (d_in[6],[12],[18]) cancel exactly through BatchNorm -> unused.

  float* ws = (float*)d_ws;
  size_t off = 0;
  const size_t NKH = (size_t)NNODES * 256;
  float* hbuf  = ws + off; off += NKH;
  float* feat  = ws + off; off += NKH;
  float* rst   = ws + off; off += NKH;
  float* sbuf  = ws + off; off += (size_t)NEDGES * 4;
  float* elb   = ws + off; off += (size_t)NNODES * 4;
  float* erb   = ws + off; off += (size_t)NNODES * 4;
  float* mbuf  = ws + off; off += (size_t)NNODES * 4;
  float* denom = ws + off; off += (size_t)NNODES * 4;
  float* bnsum = ws + off; off += 256;
  float* bnsqs = ws + off; off += 256;
  (void)ws_size; (void)in_sizes; (void)n_in;

  const int Kh[3]  = {4, 4, 1};
  const int Fh[3]  = {64, 64, 40};
  const int Kin[3] = {INDIM, 256, 256};

  const float* hcur = x;
  for (int l = 0; l < 3; ++l) {
    const int K = Kh[l], F = Fh[l], KH = K * F, Kd = Kin[l];

    dim3 gg((NNODES + 63) / 64, (KH + 127) / 128);
    gemm_wmma_bf16<<<gg, 256, 0, stream>>>(hcur, W[l], feat, NNODES, Kd, KH);

    attn_dots<<<(NNODES + 7) / 8, 256, 0, stream>>>(feat, al[l], ar[l], elb, erb, K, F);

    long nrst = (long)NNODES * KH;
    long nk   = (long)NNODES * K;
    fill_kernel<<<(unsigned)((nrst + 255) / 256), 256, 0, stream>>>(rst, 0.0f, nrst);
    fill_kernel<<<(unsigned)((nk + 255) / 256),   256, 0, stream>>>(mbuf, -INFINITY, nk);
    fill_kernel<<<(unsigned)((nk + 255) / 256),   256, 0, stream>>>(denom, 0.0f, nk);
    fill_kernel<<<1, 256, 0, stream>>>(bnsum, 0.0f, 256);
    fill_kernel<<<1, 256, 0, stream>>>(bnsqs, 0.0f, 256);

    long ek = (long)NEDGES * K;
    edge_score<<<(unsigned)((ek + 255) / 256), 256, 0, stream>>>(src, dst, elb, erb, sbuf, mbuf, K);
    edge_exp<<<(unsigned)((ek + 255) / 256), 256, 0, stream>>>(dst, mbuf, sbuf, denom, K);
    aggregate<<<(NEDGES + 7) / 8, 256, 0, stream>>>(src, dst, feat, sbuf, denom, rst, K, F);

    const int rpb = 1024;
    bn_stats<<<(NNODES + rpb - 1) / rpb, 256, 0, stream>>>(rst, bnsum, bnsqs, KH, rpb);
    bn_apply_relu<<<(unsigned)((nrst + 255) / 256), 256, 0, stream>>>(rst, bnsum, bnsqs,
                                                                      g[l], be[l], hbuf, KH);
    hcur = hbuf;
  }
  softmax40<<<(NNODES + 7) / 8, 256, 0, stream>>>(hbuf, (float*)d_out);
}
